// FullAttention_52974126629409
// MI455X (gfx1250) — matrix operand order repaired and checked by reading, not compiled, not run on hardware
//
#include <hip/hip_runtime.h>
#include <hip/hip_fp16.h>

#define DIMC   512
#define NHEAD  8
#define HD     64
#define NTOK   2048
#define NB     2

typedef _Float16 half16 __attribute__((ext_vector_type(16)));
typedef float    float8 __attribute__((ext_vector_type(8)));

union Frag { half16 h; uint4 q[2]; };
union H2   { _Float16 h[2]; unsigned u; };

__device__ __forceinline__ float8 wmma_f16(const Frag& a, const Frag& b, float8 c) {
  // D = A(16x32 f16) * B(32x16 f16) + C(16x16 f32)
  return __builtin_amdgcn_wmma_f32_16x16x32_f16(false, a.h, false, b.h,
                                                (short)0, c, false, false);
}

// A-matrix fragment from row-major f16 (row = lc for both half-waves).
// lanes 0-15: K {c*32+0..7, c*32+16..23}; lanes 16-31: K {c*32+8..15, c*32+24..31}
__device__ __forceinline__ void loadA(Frag& f, const _Float16* base, int stride,
                                      int lc, int hi, int kchunk) {
  int k0 = kchunk * 32 + hi * 8;
  const _Float16* p = base + lc * stride + k0;
  f.q[0] = *(const uint4*)(p);
  f.q[1] = *(const uint4*)(p + 16);
}

// B-matrix fragment from column-major staged f16 (Bt[col][k]).
// lane n holds column ncol0+n; lanes 0-15: K {c*32+0..7, c*32+16..23}; lanes 16-31: K {c*32+8..15, c*32+24..31}
__device__ __forceinline__ void loadB(Frag& f, const _Float16* base, int stride,
                                      int lc, int hi, int ncol0, int kchunk) {
  int k0 = kchunk * 32 + hi * 8;
  const _Float16* p = base + (ncol0 + lc) * stride + k0;
  f.q[0] = *(const uint4*)(p);
  f.q[1] = *(const uint4*)(p + 16);
}

// Async global->LDS 16-byte copy (GVS mode), ASYNCcnt-tracked; bypasses VGPRs.
__device__ __forceinline__ void async_b128(const _Float16* lds_dst,
                                           unsigned long long gbase,
                                           unsigned byte_off) {
  unsigned ldsa = (unsigned)(uintptr_t)lds_dst;   // flat LDS addr truncates to offset
  asm volatile("global_load_async_to_lds_b128 %0, %1, %2"
               :: "v"(ldsa), "v"(byte_off), "s"(gbase) : "memory");
}
__device__ __forceinline__ void wait_async0() {
  asm volatile("s_wait_asynccnt 0" ::: "memory");
}

// ---------------- Kernel 1: QKV projection ----------------------------------
// [4096,512] f32 x [512,1536] f32 + bias -> Q/K/V f16 in [b][h][n][hd] layout
__global__ __launch_bounds__(128) void qkv_kernel(const float* __restrict__ x,
                                                  const float* __restrict__ w,
                                                  const float* __restrict__ bvec,
                                                  _Float16* __restrict__ qkv) {
  __shared__ _Float16 Alds[64 * 40];
  __shared__ _Float16 Bt[64 * 40];
  const int tid = threadIdx.x;
  const int wave = tid >> 5, lane = tid & 31, lc = lane & 15, hi = lane >> 4;
  const int n0 = blockIdx.x * 64;   // 0..1535
  const int m0 = blockIdx.y * 64;   // 0..4095
  float8 acc[4] = {};
  for (int k0 = 0; k0 < DIMC; k0 += 32) {
    __syncthreads();
#pragma unroll
    for (int i = 0; i < 8; ++i) {          // A tile 64x32 f32 -> f16
      int p = tid + i * 128, r = p >> 4, c2 = p & 15;
      float2 v = *(const float2*)(x + (size_t)(m0 + r) * DIMC + k0 + c2 * 2);
      H2 h; h.h[0] = (_Float16)v.x; h.h[1] = (_Float16)v.y;
      *(unsigned*)(&Alds[r * 40 + c2 * 2]) = h.u;
    }
#pragma unroll
    for (int i = 0; i < 8; ++i) {          // B tile 32x64 -> transposed f16
      int p = tid + i * 128, r = p >> 5, c2 = p & 31;
      float2 v = *(const float2*)(w + (size_t)(k0 + r) * 1536 + n0 + c2 * 2);
      Bt[(c2 * 2) * 40 + r]     = (_Float16)v.x;
      Bt[(c2 * 2 + 1) * 40 + r] = (_Float16)v.y;
    }
    __syncthreads();
    Frag a; loadA(a, &Alds[wave * 16 * 40], 40, lc, hi, 0);
#pragma unroll
    for (int t = 0; t < 4; ++t) {
      Frag b; loadB(b, Bt, 40, lc, hi, t * 16, 0);
      acc[t] = wmma_f16(a, b, acc[t]);
    }
  }
#pragma unroll
  for (int t = 0; t < 4; ++t) {
    int gc = n0 + t * 16 + lc;              // 0..1535
    float bias = bvec[gc];
    int which = gc >> 9, head = (gc >> 6) & 7, hd = gc & 63;
    _Float16* dstbase = qkv + (size_t)which * (NB * NHEAD * NTOK * HD);
#pragma unroll
    for (int r = 0; r < 8; ++r) {
      int gr = m0 + wave * 16 + r + 8 * hi;
      int b = gr >> 11, n = gr & 2047;
      dstbase[((size_t)(b * NHEAD + head) * NTOK + n) * HD + hd] =
          (_Float16)(acc[t][r] + bias);
    }
  }
}

// ---------------- Kernel 2: flash attention ---------------------------------
__global__ __launch_bounds__(128) void attn_kernel(const _Float16* __restrict__ qkv,
                                                   _Float16* __restrict__ att) {
  __shared__ _Float16 Klds[64 * 80];
  __shared__ _Float16 Vt[64 * 80];
  __shared__ _Float16 Plds[4 * 16 * 80];
  const int tid = threadIdx.x;
  const int wave = tid >> 5, lane = tid & 31, lc = lane & 15, hi = lane >> 4;
  const int qt = blockIdx.x & 31;
  const int h  = (blockIdx.x >> 5) & 7;
  const int b  = blockIdx.x >> 8;
  const _Float16* Q = qkv + (size_t)(b * NHEAD + h) * NTOK * HD;
  const _Float16* K = Q + (size_t)NB * NHEAD * NTOK * HD;
  const _Float16* V = K + (size_t)NB * NHEAD * NTOK * HD;
  const unsigned long long kbase = (unsigned long long)K;
  const int q0 = qt * 64 + wave * 16;

  Frag aq[2];                               // resident Q fragments (16x64)
  loadA(aq[0], Q + (size_t)q0 * HD, HD, lc, hi, 0);
  loadA(aq[1], Q + (size_t)q0 * HD, HD, lc, hi, 1);

  float8 O[4] = {};
  float mrow[8], lrow[8];
#pragma unroll
  for (int r = 0; r < 8; ++r) { mrow[r] = -__builtin_huge_valf(); lrow[r] = 0.f; }
  _Float16* Pw = Plds + wave * 16 * 80;

  for (int blk = 0; blk < 32; ++blk) {
    const int m0 = blk * 64;
    __syncthreads();                        // previous-iter LDS reads done
    // K block [64][64] -> Klds via async global->LDS b128 (512 chunks of 16B)
#pragma unroll
    for (int i = 0; i < 4; ++i) {
      int p = tid + i * 128, r = p >> 3, c = p & 7;
      async_b128(&Klds[r * 80 + c * 8], kbase,
                 (unsigned)((m0 + r) * (HD * 2) + c * 16));
    }
    // V block transposed [hd][key] (manual: needs transpose), overlaps async K
#pragma unroll
    for (int i = 0; i < 16; ++i) {
      int p = tid + i * 128, r = p >> 5, c2 = p & 31;
      H2 v; v.u = *(const unsigned*)(V + (size_t)(m0 + r) * HD + c2 * 2);
      Vt[(c2 * 2) * 80 + r]     = v.h[0];
      Vt[(c2 * 2 + 1) * 80 + r] = v.h[1];
    }
    wait_async0();
    __syncthreads();

    float8 S[4];                            // S = Q K^T  (16 x 64)
#pragma unroll
    for (int t = 0; t < 4; ++t) {
      float8 s = {};
#pragma unroll
      for (int c = 0; c < 2; ++c) {
        Frag bk; loadB(bk, Klds, 80, lc, hi, t * 16, c);
        s = wmma_f16(aq[c], bk, s);
      }
      S[t] = s;
    }
#pragma unroll
    for (int t = 0; t < 4; ++t)
#pragma unroll
      for (int r = 0; r < 8; ++r)
        S[t][r] = fminf(fmaxf(S[t][r] * 0.125f, -10000.f), 10000.f);

    float mnew[8], alpha[8];
#pragma unroll
    for (int r = 0; r < 8; ++r) {           // row max across 64 keys
      float v = fmaxf(fmaxf(S[0][r], S[1][r]), fmaxf(S[2][r], S[3][r]));
      v = fmaxf(v, __shfl_xor(v, 1));
      v = fmaxf(v, __shfl_xor(v, 2));
      v = fmaxf(v, __shfl_xor(v, 4));
      v = fmaxf(v, __shfl_xor(v, 8));
      mnew[r] = fmaxf(mrow[r], v);
      alpha[r] = __expf(mrow[r] - mnew[r]);
      mrow[r] = mnew[r];
    }
#pragma unroll
    for (int r = 0; r < 8; ++r) {           // P = exp(S-m), sum, rescale O
      float rs = 0.f;
      int prow = r + 8 * hi;
#pragma unroll
      for (int t = 0; t < 4; ++t) {
        float e = __expf(S[t][r] - mnew[r]);
        rs += e;
        Pw[prow * 80 + t * 16 + lc] = (_Float16)e;   // D-layout -> plain [16][80]
      }
      rs += __shfl_xor(rs, 1);
      rs += __shfl_xor(rs, 2);
      rs += __shfl_xor(rs, 4);
      rs += __shfl_xor(rs, 8);
      lrow[r] = lrow[r] * alpha[r] + rs;
#pragma unroll
      for (int t = 0; t < 4; ++t) O[t][r] *= alpha[r];
    }

    Frag ap[2];                             // reload P in A-layout
    loadA(ap[0], Pw, 80, lc, hi, 0);
    loadA(ap[1], Pw, 80, lc, hi, 1);
#pragma unroll
    for (int t = 0; t < 4; ++t)             // O += P * V
#pragma unroll
      for (int c = 0; c < 2; ++c) {
        Frag bv; loadB(bv, Vt, 80, lc, hi, t * 16, c);
        O[t] = wmma_f16(ap[c], bv, O[t]);
      }
  }

#pragma unroll
  for (int r = 0; r < 8; ++r) {             // normalize + write [4096][512] f16
    float linv = 1.f / lrow[r];
    int q = q0 + r + 8 * hi;
    _Float16* dst = att + ((size_t)(b * NTOK + q)) * DIMC + h * HD;
#pragma unroll
    for (int t = 0; t < 4; ++t) dst[t * 16 + lc] = (_Float16)(O[t][r] * linv);
  }
}

// ---------------- Kernel 3: output projection -------------------------------
__global__ __launch_bounds__(128) void proj_kernel(const _Float16* __restrict__ att,
                                                   const float* __restrict__ w,
                                                   const float* __restrict__ bvec,
                                                   float* __restrict__ out) {
  __shared__ _Float16 Alds[64 * 40];
  __shared__ _Float16 Bt[64 * 40];
  const int tid = threadIdx.x;
  const int wave = tid >> 5, lane = tid & 31, lc = lane & 15, hi = lane >> 4;
  const int n0 = blockIdx.x * 64;   // 0..511
  const int m0 = blockIdx.y * 64;   // 0..4095
  const unsigned long long abase = (unsigned long long)att;
  float8 acc[4] = {};
  for (int k0 = 0; k0 < DIMC; k0 += 32) {
    __syncthreads();
    // A tile already f16: async global->LDS b128 (256 chunks of 16B)
#pragma unroll
    for (int i = 0; i < 2; ++i) {
      int p = tid + i * 128, r = p >> 2, c = p & 3;
      async_b128(&Alds[r * 40 + c * 8], abase,
                 (unsigned)((m0 + r) * (DIMC * 2) + k0 * 2 + c * 16));
    }
#pragma unroll
    for (int i = 0; i < 8; ++i) {           // B tile 32x64 f32 -> transposed f16
      int p = tid + i * 128, r = p >> 5, c2 = p & 31;
      float2 v = *(const float2*)(w + (size_t)(k0 + r) * DIMC + n0 + c2 * 2);
      Bt[(c2 * 2) * 40 + r]     = (_Float16)v.x;
      Bt[(c2 * 2 + 1) * 40 + r] = (_Float16)v.y;
    }
    wait_async0();
    __syncthreads();
    Frag a; loadA(a, &Alds[wave * 16 * 40], 40, lc, hi, 0);
#pragma unroll
    for (int t = 0; t < 4; ++t) {
      Frag b; loadB(b, Bt, 40, lc, hi, t * 16, 0);
      acc[t] = wmma_f16(a, b, acc[t]);
    }
  }
#pragma unroll
  for (int t = 0; t < 4; ++t) {
    int gc = n0 + t * 16 + lc;
    float bias = bvec[gc];
#pragma unroll
    for (int r = 0; r < 8; ++r) {
      int gr = m0 + wave * 16 + r + 8 * hi;
      out[(size_t)gr * DIMC + gc] = acc[t][r] + bias;
    }
  }
}

extern "C" void kernel_launch(void* const* d_in, const int* in_sizes, int n_in,
                              void* d_out, int out_size, void* d_ws, size_t ws_size,
                              hipStream_t stream) {
  const float* x      = (const float*)d_in[0];
  const float* qkv_w  = (const float*)d_in[1];
  const float* qkv_b  = (const float*)d_in[2];
  const float* proj_w = (const float*)d_in[3];
  const float* proj_b = (const float*)d_in[4];

  _Float16* qkv = (_Float16*)d_ws;                              // 3 x 4 MB f16
  _Float16* att = qkv + (size_t)3 * NB * NHEAD * NTOK * HD;     // 4 MB f16

  // QKV: [4096,1536] output in 64x64 tiles
  qkv_kernel<<<dim3(1536 / 64, 4096 / 64), 128, 0, stream>>>(x, qkv_w, qkv_b, qkv);
  // Attention: B*H*(N/64) = 512 workgroups, 4 waves each
  attn_kernel<<<dim3(NB * NHEAD * (NTOK / 64)), 128, 0, stream>>>(qkv, att);
  // Projection: [4096,512] output
  proj_kernel<<<dim3(DIMC / 64, 4096 / 64), 128, 0, stream>>>(att, proj_w, proj_b,
                                                              (float*)d_out);
}
